// Attention_23252952940614
// MI455X (gfx1250) — matrix operand order repaired and checked by reading, not compiled, not run on hardware
//
#include <hip/hip_runtime.h>
#include <hip/hip_bf16.h>
#include <math.h>

typedef __bf16 bf16;
typedef __attribute__((ext_vector_type(16))) __bf16 v16bf;
typedef __attribute__((ext_vector_type(8)))  __bf16 v8bf;
typedef __attribute__((ext_vector_type(8)))  float  v8f;

// ---------------- WMMA helpers (CDNA5 16x16x32 bf16, f32 accum) ----------------

static __device__ inline v8f wmma_bf16(v16bf a, v16bf b, v8f c) {
  // 8 args: (neg_a, A, neg_b, B, c_mod, C, reuse_a, reuse_b)
  return __builtin_amdgcn_wmma_f32_16x16x32_bf16(false, a, false, b, (short)0, c,
                                                 false, false);
}

// A-matrix 16x32 bf16 fragment (ISA 7.12.2): lane M = lane&15;
// elements 0..7  -> K = kbase + e        (kbase = (lane>>4)*8)
// elements 8..15 -> K = 16 + kbase + (e-8)
static __device__ inline v16bf load_a16(const bf16* __restrict__ base, int ld,
                                        int row0, int k0) {
  const int lane = threadIdx.x & 31;
  const bf16* p = base + (size_t)(row0 + (lane & 15)) * ld + k0 + ((lane >> 4) << 3);
  v8bf lo = *(const v8bf*)p;
  v8bf hi = *(const v8bf*)(p + 16);
  v16bf r;
#pragma unroll
  for (int i = 0; i < 8; ++i) { r[i] = lo[i]; r[i + 8] = hi[i]; }
  return r;
}

// B-matrix 32x16 bf16 fragment: lane column N = lane&15;
// elements 0..7 -> K = (lane>>4)*8 + e; elements 8..15 -> K = 16 + (lane>>4)*8 + (e-8)
static __device__ inline v16bf load_b16(const bf16* __restrict__ base, int ld,
                                        int col0, int k0) {
  const int lane = threadIdx.x & 31;
  const bf16* p = base + (size_t)(col0 + (lane & 15)) * ld + k0 + ((lane >> 4) << 3);
  v8bf lo = *(const v8bf*)p;
  v8bf hi = *(const v8bf*)(p + 16);
  v16bf r;
#pragma unroll
  for (int i = 0; i < 8; ++i) { r[i] = lo[i]; r[i + 8] = hi[i]; }
  return r;
}

// ---------------- elementwise kernels ----------------

__global__ void cvt_bf16(const float* __restrict__ in, bf16* __restrict__ out, int n) {
  int i = blockIdx.x * blockDim.x + threadIdx.x;
  if (i < n) out[i] = (bf16)in[i];
}

// qraw: [B*S, H*HD] bf16 -> Qb: [B,H,S,HD] bf16 with RoPE applied
__global__ void rope_q(const bf16* __restrict__ qraw, const float* __restrict__ cosb,
                       const float* __restrict__ sinb, bf16* __restrict__ Qb) {
  const int HD = 128, H = 16, S = 2048;
  int idx = blockIdx.x * blockDim.x + threadIdx.x;   // B*S*H*HD
  int d = idx & (HD - 1);
  int h = (idx >> 7) & (H - 1);
  int s = (idx >> 11) & (S - 1);
  int b = idx >> 22;
  size_t base = (((size_t)b * S + s) * H + h) * HD;
  float q = (float)qraw[base + d];
  float rot = (d < 64) ? -(float)qraw[base + d + 64] : (float)qraw[base + d - 64];
  float o = q * cosb[s * HD + d] + rot * sinb[s * HD + d];
  Qb[((((size_t)b * H + h) * S) + s) * HD + d] = (bf16)o;
}

// kraw: [B*S, HKV*HD] -> Kb: [B,HKV,S,HD] with RoPE
__global__ void rope_k(const bf16* __restrict__ kraw, const float* __restrict__ cosb,
                       const float* __restrict__ sinb, bf16* __restrict__ Kb) {
  const int HD = 128, HKV = 4, S = 2048;
  int idx = blockIdx.x * blockDim.x + threadIdx.x;   // B*S*HKV*HD
  int d = idx & (HD - 1);
  int h = (idx >> 7) & (HKV - 1);
  int s = (idx >> 9) & (S - 1);
  int b = idx >> 20;
  size_t base = (((size_t)b * S + s) * HKV + h) * HD;
  float k = (float)kraw[base + d];
  float rot = (d < 64) ? -(float)kraw[base + d + 64] : (float)kraw[base + d - 64];
  float o = k * cosb[s * HD + d] + rot * sinb[s * HD + d];
  Kb[((((size_t)b * HKV + h) * S) + s) * HD + d] = (bf16)o;
}

// vraw: [B*S, HKV*HD] -> Vt: [B,HKV,HD,S] (transposed so PV B-frags are contiguous)
__global__ void pack_vt(const bf16* __restrict__ vraw, bf16* __restrict__ Vt) {
  const int HD = 128, HKV = 4, S = 2048;
  int idx = blockIdx.x * blockDim.x + threadIdx.x;   // B*S*HKV*HD
  int d = idx & (HD - 1);
  int h = (idx >> 7) & (HKV - 1);
  int s = (idx >> 9) & (S - 1);
  int b = idx >> 20;
  bf16 v = vraw[(((size_t)b * S + s) * HKV + h) * HD + d];
  Vt[((((size_t)b * HKV + h) * HD) + d) * S + s] = v;
}

// ---- GEMM: Y[M,N] = A[M,K] * W[N,K]^T, one wave = 64x64 tile (16 WMMA / k-step) ----

template <typename OutT>
__global__ __launch_bounds__(256) void gemm_xwT(const bf16* __restrict__ A,
                                                const bf16* __restrict__ W,
                                                OutT* __restrict__ Y,
                                                int M, int N, int K) {
  const int wid = threadIdx.x >> 5;
  const int gw = blockIdx.x * 8 + wid;
  const int tilesN = N >> 6;
  const int tM = (gw / tilesN) << 6;   // 64 rows / wave
  const int tN = (gw % tilesN) << 6;   // 64 cols / wave
  if (tM >= M) return;
  const int lane = threadIdx.x & 31;

  v8f acc[4][4] = {};
  for (int k = 0; k < K; k += 32) {
    v16bf a[4], b[4];
#pragma unroll
    for (int i = 0; i < 4; ++i) a[i] = load_a16(A, K, tM + i * 16, k);
#pragma unroll
    for (int j = 0; j < 4; ++j) b[j] = load_b16(W, K, tN + j * 16, k);
#pragma unroll
    for (int i = 0; i < 4; ++i)
#pragma unroll
      for (int j = 0; j < 4; ++j)
        acc[i][j] = wmma_bf16(a[i], b[j], acc[i][j]);
  }
  const int rbase = (lane >> 4) << 3;
  const int c0 = lane & 15;
#pragma unroll
  for (int i = 0; i < 4; ++i)
#pragma unroll
    for (int j = 0; j < 4; ++j)
#pragma unroll
      for (int e = 0; e < 8; ++e)
        Y[(size_t)(tM + i * 16 + rbase + e) * N + tN + j * 16 + c0] =
            (OutT)acc[i][j][e];
}

// ---------------- causal flash attention: one wave per 16-row Q tile ----------------

__global__ __launch_bounds__(256) void flash_attn(const bf16* __restrict__ Qb,
                                                  const bf16* __restrict__ Kb,
                                                  const bf16* __restrict__ Vt,
                                                  bf16* __restrict__ attn) {
  const int S = 2048, HD = 128, H = 16, HKV = 4, NREP = 4;
  __shared__ bf16 lds[8][16][32];          // per-wave P staging (8KB total)
  const int wid = threadIdx.x >> 5;
  const int lane = threadIdx.x & 31;
  const int gw = blockIdx.x * 8 + wid;
  const int qtiles = S / 16;
  const int bh = gw / qtiles;
  const int qt = gw % qtiles;
  const int b = bh / H, h = bh % H, hk = h / NREP;
  const int qbase = qt * 16;

  const bf16* Qp = Qb + ((size_t)b * H + h) * S * HD;
  const bf16* Kp = Kb + ((size_t)b * HKV + hk) * S * HD;
  const bf16* Vp = Vt + ((size_t)b * HKV + hk) * HD * S;

  v16bf qf[4];
#pragma unroll
  for (int kk = 0; kk < 4; ++kk) qf[kk] = load_a16(Qp, HD, qbase, kk * 32);

  v8f acc[8] = {};
  float m[8], l[8];
#pragma unroll
  for (int e = 0; e < 8; ++e) { m[e] = -3.0e38f; l[e] = 0.f; }

  const float scale = 0.08838834764831845f;   // 1/sqrt(128)
  const int nkv = (qbase + 16 + 31) >> 5;     // 32-wide KV tiles covering causal span
  const int rbase = qbase + ((lane >> 4) << 3);

  for (int jt = 0; jt < nkv; ++jt) {
    const int kvb = jt * 32;
    v8f s0 = {}, s1 = {};
#pragma unroll
    for (int kk = 0; kk < 4; ++kk) {
      v16bf b0 = load_b16(Kp, HD, kvb, kk * 32);
      v16bf b1 = load_b16(Kp, HD, kvb + 16, kk * 32);
      s0 = wmma_bf16(qf[kk], b0, s0);
      s1 = wmma_bf16(qf[kk], b1, s1);
    }
    const int col0 = kvb + (lane & 15);
    const int col1 = col0 + 16;
    const bool maskTile = (kvb + 31 > qbase);
    float p0[8], p1[8];
#pragma unroll
    for (int e = 0; e < 8; ++e) {
      float a0 = s0[e] * scale;
      float a1 = s1[e] * scale;
      const bool m0 = maskTile && (col0 > rbase + e);
      const bool m1 = maskTile && (col1 > rbase + e);
      if (m0) a0 = -3.0e38f;
      if (m1) a1 = -3.0e38f;
      float rmax = fmaxf(a0, a1);
#pragma unroll
      for (int w = 1; w < 16; w <<= 1) rmax = fmaxf(rmax, __shfl_xor(rmax, w, 32));
      const float mn = fmaxf(m[e], rmax);
      const float e0 = m0 ? 0.f : __expf(a0 - mn);
      const float e1 = m1 ? 0.f : __expf(a1 - mn);
      float rs = e0 + e1;
#pragma unroll
      for (int w = 1; w < 16; w <<= 1) rs += __shfl_xor(rs, w, 32);
      const float corr = __expf(m[e] - mn);
      l[e] = l[e] * corr + rs;
      m[e] = mn;
      p0[e] = e0;
      p1[e] = e1;
#pragma unroll
      for (int n = 0; n < 8; ++n) acc[n][e] *= corr;
    }
    // stage P (C-layout) to LDS, reload in A-layout
#pragma unroll
    for (int e = 0; e < 8; ++e) {
      lds[wid][((lane >> 4) << 3) + e][lane & 15] = (bf16)p0[e];
      lds[wid][((lane >> 4) << 3) + e][16 + (lane & 15)] = (bf16)p1[e];
    }
    asm volatile("s_wait_dscnt 0" ::: "memory");   // same-wave LDS RAW fence
    v16bf pf = load_a16(&lds[wid][0][0], 32, 0, 0);
#pragma unroll
    for (int n = 0; n < 8; ++n) {
      v16bf vb = load_b16(Vp, S, n * 16, kvb);    // Vt rows = head dims, K along S
      acc[n] = wmma_bf16(pf, vb, acc[n]);
    }
  }

  // normalize and write attn[b, s, h*HD + col]
  const int rb = (lane >> 4) << 3;
#pragma unroll
  for (int n = 0; n < 8; ++n)
#pragma unroll
    for (int e = 0; e < 8; ++e) {
      size_t row = (size_t)b * S + qbase + rb + e;
      attn[row * (H * HD) + h * HD + n * 16 + (lane & 15)] = (bf16)(acc[n][e] / l[e]);
    }
}

// ---------------- host launch ----------------

extern "C" void kernel_launch(void* const* d_in, const int* in_sizes, int n_in,
                              void* d_out, int out_size, void* d_ws, size_t ws_size,
                              hipStream_t stream) {
  const int B = 2, S = 2048, HID = 2048, H = 16, HKV = 4, HD = 128;
  const int M = B * S;  // 4096

  const float* x    = (const float*)d_in[0];
  const float* cosb = (const float*)d_in[1];
  const float* sinb = (const float*)d_in[2];
  const float* wq   = (const float*)d_in[3];
  const float* wk   = (const float*)d_in[4];
  const float* wv   = (const float*)d_in[5];
  const float* wo   = (const float*)d_in[6];
  float* out = (float*)d_out;

  char* ws = (char*)d_ws;
  size_t off = 0;
  auto alloc = [&](size_t bytes) -> char* {
    char* p = ws + off;
    off += (bytes + 255) & ~(size_t)255;
    return p;
  };

  bf16* xb   = (bf16*)alloc((size_t)M * HID * 2);
  bf16* wqb  = (bf16*)alloc((size_t)H * HD * HID * 2);
  bf16* wkb  = (bf16*)alloc((size_t)HKV * HD * HID * 2);
  bf16* wvb  = (bf16*)alloc((size_t)HKV * HD * HID * 2);
  bf16* wob  = (bf16*)alloc((size_t)HID * H * HD * 2);
  bf16* qraw = (bf16*)alloc((size_t)M * H * HD * 2);   // aliased as attn later
  bf16* kraw = (bf16*)alloc((size_t)M * HKV * HD * 2);
  bf16* vraw = (bf16*)alloc((size_t)M * HKV * HD * 2);
  bf16* Qbuf = (bf16*)alloc((size_t)B * H * S * HD * 2);
  bf16* Kbuf = (bf16*)alloc((size_t)B * HKV * S * HD * 2);
  bf16* Vtb  = (bf16*)alloc((size_t)B * HKV * HD * S * 2);
  bf16* attn = qraw;  // qraw fully consumed by rope_q before flash_attn writes

  // 1) f32 -> bf16 conversions
  {
    int n;
    n = M * HID;          cvt_bf16<<<(n + 255) / 256, 256, 0, stream>>>(x, xb, n);
    n = H * HD * HID;     cvt_bf16<<<(n + 255) / 256, 256, 0, stream>>>(wq, wqb, n);
    n = HKV * HD * HID;   cvt_bf16<<<(n + 255) / 256, 256, 0, stream>>>(wk, wkb, n);
    n = HKV * HD * HID;   cvt_bf16<<<(n + 255) / 256, 256, 0, stream>>>(wv, wvb, n);
    n = HID * H * HD;     cvt_bf16<<<(n + 255) / 256, 256, 0, stream>>>(wo, wob, n);
  }

  // 2) QKV projections (WMMA GEMMs, 64x64 wave tiles), bf16 outputs
  gemm_xwT<bf16><<<(M / 64) * ((H * HD) / 64) / 8, 256, 0, stream>>>(
      xb, wqb, qraw, M, H * HD, HID);
  gemm_xwT<bf16><<<(M / 64) * ((HKV * HD) / 64) / 8, 256, 0, stream>>>(
      xb, wkb, kraw, M, HKV * HD, HID);
  gemm_xwT<bf16><<<(M / 64) * ((HKV * HD) / 64) / 8, 256, 0, stream>>>(
      xb, wvb, vraw, M, HKV * HD, HID);

  // 3) RoPE + layout packs
  rope_q<<<(B * S * H * HD) / 256, 256, 0, stream>>>(qraw, cosb, sinb, Qbuf);
  rope_k<<<(B * S * HKV * HD) / 256, 256, 0, stream>>>(kraw, cosb, sinb, Kbuf);
  pack_vt<<<(B * S * HKV * HD) / 256, 256, 0, stream>>>(vraw, Vtb);

  // 4) causal flash attention (WMMA for QK^T and PV)
  flash_attn<<<(B * H * (S / 16)) / 8, 256, 0, stream>>>(Qbuf, Kbuf, Vtb, attn);

  // 5) output projection, f32 result to d_out
  gemm_xwT<float><<<(M / 64) * (HID / 64) / 8, 256, 0, stream>>>(
      attn, wob, out, M, HID, H * HD);
}
